// HypformerAttention_43937515438612
// MI455X (gfx1250) — matrix operand order repaired and checked by reading, not compiled, not run on hardware
//
#include <hip/hip_runtime.h>
#include <stdint.h>

// ---------------------------------------------------------------------------
// Hyperbolic (Lorentz) multi-head attention for MI455X (gfx1250, wave32).
// Compute-bound (~114 GFLOP, working set < 80MB -> lives in 192MB L2).
// Strategy: f16 WMMA (v_wmma_f32_16x16x32_f16) for projections, QK^T and P*V,
// flash-attention streaming over keys, Lorentz -t_q*t_k folded into padded dim.
// ---------------------------------------------------------------------------

#define NTOK  4096
#define HEADS 12
#define IN_S  256
#define OUT_S 128
#define DPAD  160   // q/k row padded to 5 chunks of K=32 (d=128 holds t, rest 0)
#define VROWS 144   // V^T rows padded to 9 tiles of 16 (row 128 holds t, rest 0)
#define DOUT  129

typedef _Float16 half_t;
typedef _Float16 v8h  __attribute__((ext_vector_type(8)));
typedef _Float16 v16h __attribute__((ext_vector_type(16)));
typedef float    v8f  __attribute__((ext_vector_type(8)));

__device__ __forceinline__ v8f wmma16x16x32(v16h a, v16h b, v8f c) {
  // D = A(16x32,f16) x B(32x16,f16) + C(16x16,f32)
  return __builtin_amdgcn_wmma_f32_16x16x32_f16(false, a, false, b, (short)0, c,
                                                false, false);
}

// A-fragment (16x32 f16): lane (p = lane%16, h2 = lane/16) holds row p,
// k = {8*h2 .. 8*h2+7} U {16+8*h2 .. 16+8*h2+7}  -> two 16B loads.
__device__ __forceinline__ v16h afrag(const half_t* rowk0, int h2) {
  v8h lo = *(const v8h*)(rowk0 + 8 * h2);
  v8h hi = *(const v8h*)(rowk0 + 16 + 8 * h2);
  return __builtin_shufflevector(lo, hi, 0, 1, 2, 3, 4, 5, 6, 7, 8, 9, 10, 11,
                                 12, 13, 14, 15);
}
// B-fragment (32x16 f16): lane (p,h2) holds column p,
// k = {8*h2 .. 8*h2+7} U {16+8*h2 .. 16+8*h2+7}  -> two 16B loads.
__device__ __forceinline__ v16h bfrag(const half_t* colk0, int h2) {
  v8h lo = *(const v8h*)(colk0 + 8 * h2);
  v8h hi = *(const v8h*)(colk0 + 16 + 8 * h2);
  return __builtin_shufflevector(lo, hi, 0, 1, 2, 3, 4, 5, 6, 7, 8, 9, 10, 11,
                                 12, 13, 14, 15);
}

// Cross-lane reductions via ds_swizzle (group-of-32, and_mask=0x1F, xor in [14:10]).
#define SWZ_F(v, imm) \
  __int_as_float(__builtin_amdgcn_ds_swizzle(__float_as_int(v), (imm)))
__device__ __forceinline__ float rmax16(float v) {  // within 16-lane half
  v = fmaxf(v, SWZ_F(v, (1 << 10) | 0x1F));
  v = fmaxf(v, SWZ_F(v, (2 << 10) | 0x1F));
  v = fmaxf(v, SWZ_F(v, (4 << 10) | 0x1F));
  v = fmaxf(v, SWZ_F(v, (8 << 10) | 0x1F));
  return v;
}
__device__ __forceinline__ float rsum16(float v) {
  v += SWZ_F(v, (1 << 10) | 0x1F);
  v += SWZ_F(v, (2 << 10) | 0x1F);
  v += SWZ_F(v, (4 << 10) | 0x1F);
  v += SWZ_F(v, (8 << 10) | 0x1F);
  return v;
}
__device__ __forceinline__ float rsum32(float v) {
  v = rsum16(v);
  v += SWZ_F(v, (16 << 10) | 0x1F);
  return v;
}

// ---------------------------------------------------------------------------
__global__ void k_zero(uint32_t* __restrict__ p, unsigned count) {
  unsigned i = blockIdx.x * blockDim.x + threadIdx.x;
  unsigned stride = gridDim.x * blockDim.x;
  for (; i < count; i += stride) p[i] = 0u;
}

__global__ void k_convert(const float* __restrict__ x, const float* __restrict__ Wq,
                          const float* __restrict__ Wk, const float* __restrict__ Wv,
                          half_t* __restrict__ xs, half_t* __restrict__ wq,
                          half_t* __restrict__ wk, half_t* __restrict__ wv) {
  unsigned i = blockIdx.x * blockDim.x + threadIdx.x;
  unsigned stride = gridDim.x * blockDim.x;
  const unsigned NX = NTOK * IN_S;
  const unsigned NW = HEADS * OUT_S * IN_S;
  for (unsigned k = i; k < NX; k += stride) {
    unsigned row = k / IN_S, col = k % IN_S;
    xs[k] = (half_t)x[row * (IN_S + 1) + 1 + col];  // drop lorentz time column
  }
  for (unsigned k = i; k < NW; k += stride) {
    wq[k] = (half_t)Wq[k];
    wk[k] = (half_t)Wk[k];
    wv[k] = (half_t)Wv[k];
  }
}

// ---------------------------------------------------------------------------
// QKV projection: per head, s[n,o] = xs[n,:] . W[h,o,:] + b, t = sqrt(|s|^2+1).
// grid = (N/64, HEADS, 3 projections), block = 128 (4 waves x 16 rows).
__global__ void __launch_bounds__(128)
k_proj(const half_t* __restrict__ xs, const half_t* __restrict__ w16q,
       const half_t* __restrict__ w16k, const half_t* __restrict__ w16v,
       const float* __restrict__ bq, const float* __restrict__ bk,
       const float* __restrict__ bv, half_t* __restrict__ Qh,
       half_t* __restrict__ Kh, half_t* __restrict__ Vt) {
  const int lane = threadIdx.x & 31;
  const int wave = threadIdx.x >> 5;
  const int p = lane & 15, h2 = lane >> 4;
  const int head = blockIdx.y;
  const int proj = blockIdx.z;
  const int rowbase = blockIdx.x * 64 + wave * 16;

  const half_t* w16 = (proj == 0) ? w16q : (proj == 1) ? w16k : w16v;
  const float* bb = (proj == 0) ? bq : (proj == 1) ? bk : bv;

  const half_t* xrow = xs + (size_t)(rowbase + p) * IN_S;
  v16h af[8];
#pragma unroll
  for (int kc = 0; kc < 8; ++kc) af[kc] = afrag(xrow + kc * 32, h2);

  float part[8];
#pragma unroll
  for (int r = 0; r < 8; ++r) part[r] = 0.f;

#pragma unroll
  for (int t = 0; t < 8; ++t) {
    v8f c = {0, 0, 0, 0, 0, 0, 0, 0};
    const half_t* wcol = w16 + ((size_t)head * OUT_S + (t * 16 + p)) * IN_S;
#pragma unroll
    for (int kc = 0; kc < 8; ++kc)
      c = wmma16x16x32(af[kc], bfrag(wcol + kc * 32, h2), c);
    float bvv = bb[head * OUT_S + t * 16 + p];
#pragma unroll
    for (int r = 0; r < 8; ++r) {
      float s = c[r] + bvv;
      part[r] += s * s;
      int row = rowbase + r + 8 * h2;
      int d = t * 16 + p;
      if (proj == 0)
        Qh[((size_t)head * NTOK + row) * DPAD + d] = (half_t)s;
      else if (proj == 1)
        Kh[((size_t)head * NTOK + row) * DPAD + d] = (half_t)s;
      else
        Vt[((size_t)head * VROWS + d) * NTOK + row] = (half_t)s;
    }
  }
#pragma unroll
  for (int r = 0; r < 8; ++r) {
    float tt = sqrtf(rsum16(part[r]) + 1.0f);
    if (p == r) {
      int row = rowbase + r + 8 * h2;
      if (proj == 0)
        Qh[((size_t)head * NTOK + row) * DPAD + 128] = (half_t)tt;
      else if (proj == 1)
        Kh[((size_t)head * NTOK + row) * DPAD + 128] = (half_t)(-tt);  // fold sgn
      else
        Vt[((size_t)head * VROWS + 128) * NTOK + row] = (half_t)tt;
    }
  }
}

// ---------------------------------------------------------------------------
// Flash attention. grid = (N/64, HEADS), block = 128. Each wave owns 16 query
// rows; streams 4096 keys in chunks of 32 (two 16-key score tiles).
__global__ void __launch_bounds__(128)
k_attn(const half_t* __restrict__ Qh, const half_t* __restrict__ Kh,
       const half_t* __restrict__ Vt, float* __restrict__ ave,
       const float* __restrict__ scale_p, const float* __restrict__ bias_p) {
  __shared__ half_t lds[4 * 16 * 32];  // per-wave 16x32 P tile for transpose
  const int lane = threadIdx.x & 31;
  const int wave = threadIdx.x >> 5;
  const int p = lane & 15, h2 = lane >> 4;
  const int head = blockIdx.y;
  const int qbase = blockIdx.x * 64 + wave * 16;
  const float inv_scale = 1.0f / scale_p[0];
  const float bias = bias_p[0];

  const half_t* qrow = Qh + ((size_t)head * NTOK + qbase + p) * DPAD;
  v16h qf[5];
#pragma unroll
  for (int kc = 0; kc < 5; ++kc) qf[kc] = afrag(qrow + kc * 32, h2);

  v8f o[9];
#pragma unroll
  for (int t = 0; t < 9; ++t) o[t] = (v8f){0, 0, 0, 0, 0, 0, 0, 0};
  float m[8], l[8];
#pragma unroll
  for (int r = 0; r < 8; ++r) { m[r] = -1e30f; l[r] = 0.f; }

  half_t* myP = lds + wave * (16 * 32);

  for (int kb = 0; kb < NTOK / 32; ++kb) {
    // scores: S = Q x K^T over 160 padded dims (spatial 128 + (-t_q t_k) slot)
    const half_t* kcol0 = Kh + ((size_t)head * NTOK + kb * 32 + p) * DPAD;
    const half_t* kcol1 = kcol0 + 16 * DPAD;
    v8f c0 = (v8f){0, 0, 0, 0, 0, 0, 0, 0};
    v8f c1 = (v8f){0, 0, 0, 0, 0, 0, 0, 0};
#pragma unroll
    for (int kc = 0; kc < 5; ++kc) {
      c0 = wmma16x16x32(qf[kc], bfrag(kcol0 + kc * 32, h2), c0);
      c1 = wmma16x16x32(qf[kc], bfrag(kcol1 + kc * 32, h2), c1);
    }
    // online softmax per row (row = r + 8*h2, cols across 16 lanes of a half)
#pragma unroll
    for (int r = 0; r < 8; ++r) {
      float s0 = (2.0f + 2.0f * c0[r]) * inv_scale + bias;
      float s1 = (2.0f + 2.0f * c1[r]) * inv_scale + bias;
      float mx = rmax16(fmaxf(s0, s1));
      float mnew = fmaxf(m[r], mx);
      float alpha = __expf(m[r] - mnew);
      m[r] = mnew;
      float p0 = __expf(s0 - mnew);
      float p1 = __expf(s1 - mnew);
      l[r] = l[r] * alpha + rsum16(p0 + p1);
#pragma unroll
      for (int t = 0; t < 9; ++t) o[t][r] *= alpha;
      int row = r + 8 * h2;
      myP[row * 32 + p] = (half_t)p0;
      myP[row * 32 + p + 16] = (half_t)p1;
    }
    __syncthreads();
    // P (C-layout) -> A-fragment via LDS, then O += P x V^T-chunk (9 d-tiles)
    v16h pf = afrag(myP + p * 32, h2);
#pragma unroll
    for (int t = 0; t < 9; ++t) {
      const half_t* vcol =
          Vt + ((size_t)head * VROWS + t * 16 + p) * NTOK + kb * 32;
      o[t] = wmma16x16x32(pf, bfrag(vcol, h2), o[t]);
    }
    __syncthreads();
  }

#pragma unroll
  for (int r = 0; r < 8; ++r) {
    float inv_l = 1.0f / l[r];
    int row = qbase + r + 8 * h2;
#pragma unroll
    for (int t = 0; t < 9; ++t) {
      int d = t * 16 + p;
      if (d < DOUT)
        ave[((size_t)head * NTOK + row) * DOUT + d] = o[t][r] * inv_l;
    }
  }
}

// ---------------------------------------------------------------------------
// Midpoint-norm per head, mean over heads, norm again. One wave per token.
__global__ void __launch_bounds__(256)
k_finalize(const float* __restrict__ ave, float* __restrict__ out) {
  const int lane = threadIdx.x & 31;
  const int wave = threadIdx.x >> 5;
  const int n = blockIdx.x * 8 + wave;
  float acc[5] = {0.f, 0.f, 0.f, 0.f, 0.f};
  for (int h = 0; h < HEADS; ++h) {
    const float* a = ave + ((size_t)h * NTOK + n) * DOUT;
    float v[5];
    float part = 0.f;
#pragma unroll
    for (int j = 0; j < 5; ++j) {
      int d = lane + 32 * j;
      float x = (d < DOUT) ? a[d] : 0.f;
      v[j] = x;
      part += (d == 0) ? -x * x : x * x;
    }
    float denom = sqrtf(fmaxf(fabsf(rsum32(part)), 1e-8f));
    float rinv = 1.0f / denom;
#pragma unroll
    for (int j = 0; j < 5; ++j) acc[j] += v[j] * rinv;
  }
#pragma unroll
  for (int j = 0; j < 5; ++j) acc[j] *= (1.0f / HEADS);
  float part2 = 0.f;
#pragma unroll
  for (int j = 0; j < 5; ++j) {
    int d = lane + 32 * j;
    part2 += (d == 0) ? -acc[j] * acc[j] : acc[j] * acc[j];
  }
  float rinv2 = 1.0f / sqrtf(fmaxf(fabsf(rsum32(part2)), 1e-8f));
#pragma unroll
  for (int j = 0; j < 5; ++j) {
    int d = lane + 32 * j;
    if (d < DOUT) out[(size_t)n * DOUT + d] = acc[j] * rinv2;
  }
}

// ---------------------------------------------------------------------------
extern "C" void kernel_launch(void* const* d_in, const int* in_sizes, int n_in,
                              void* d_out, int out_size, void* d_ws,
                              size_t ws_size, hipStream_t stream) {
  (void)in_sizes; (void)n_in; (void)out_size; (void)ws_size;
  const float* x = (const float*)d_in[0];
  const float* Wq = (const float*)d_in[1];
  const float* bq = (const float*)d_in[2];
  const float* Wk = (const float*)d_in[3];
  const float* bk = (const float*)d_in[4];
  const float* Wv = (const float*)d_in[5];
  const float* bv = (const float*)d_in[6];
  const float* scale = (const float*)d_in[7];
  const float* bias = (const float*)d_in[8];
  float* out = (float*)d_out;

  char* ws = (char*)d_ws;
  size_t off = 0;
  auto alloc = [&](size_t bytes) {
    char* ptr = ws + off;
    off += (bytes + 255) & ~(size_t)255;
    return ptr;
  };
  half_t* xs   = (half_t*)alloc((size_t)NTOK * IN_S * 2);           // 2.0 MB
  half_t* wq16 = (half_t*)alloc((size_t)HEADS * OUT_S * IN_S * 2);  // 0.75 MB
  half_t* wk16 = (half_t*)alloc((size_t)HEADS * OUT_S * IN_S * 2);
  half_t* wv16 = (half_t*)alloc((size_t)HEADS * OUT_S * IN_S * 2);
  half_t* Qh   = (half_t*)alloc((size_t)HEADS * NTOK * DPAD * 2);   // 15.7 MB
  half_t* Kh   = (half_t*)alloc((size_t)HEADS * NTOK * DPAD * 2);   // 15.7 MB
  half_t* Vt   = (half_t*)alloc((size_t)HEADS * VROWS * NTOK * 2);  // 14.2 MB
  float*  ave  = (float*)alloc((size_t)HEADS * NTOK * DOUT * 4);    // 25.4 MB

  // zero Qh/Kh/Vt (contiguous, 256B-multiple sizes) so pad lanes are exact 0
  unsigned zcount =
      (unsigned)(((size_t)HEADS * NTOK * DPAD * 2 * 2 +
                  (size_t)HEADS * VROWS * NTOK * 2) / 4);
  k_zero<<<dim3(2048), dim3(256), 0, stream>>>((uint32_t*)Qh, zcount);
  k_convert<<<dim3(2048), dim3(256), 0, stream>>>(x, Wq, Wk, Wv, xs, wq16,
                                                  wk16, wv16);
  k_proj<<<dim3(NTOK / 64, HEADS, 3), dim3(128), 0, stream>>>(
      xs, wq16, wk16, wv16, bq, bk, bv, Qh, Kh, Vt);
  k_attn<<<dim3(NTOK / 64, HEADS), dim3(128), 0, stream>>>(Qh, Kh, Vt, ave,
                                                           scale, bias);
  k_finalize<<<dim3(NTOK / 8), dim3(256), 0, stream>>>(ave, out);
}
